// Base_31095563223209
// MI455X (gfx1250) — matrix operand order repaired and checked by reading, not compiled, not run on hardware
//
#include <hip/hip_runtime.h>

// ---------- types ----------
typedef __attribute__((ext_vector_type(16))) __bf16         bf16x16;
typedef __attribute__((ext_vector_type(8)))  float          floatx8;
typedef __attribute__((ext_vector_type(8)))  unsigned short ushort8;

struct Bf16Pack { ushort8 lo, hi; };

// ---------- constants (match reference) ----------
#define N_NODES 50000
#define N_EDGES 200000

// fp32 -> bf16, round-to-nearest-even
__device__ __forceinline__ unsigned short f2bf(float f) {
    unsigned int u = __builtin_bit_cast(unsigned int, f);
    unsigned int r = u + 0x7FFFu + ((u >> 16) & 1u);
    return (unsigned short)(r >> 16);
}

// ---------- CDNA5 async copy: global -> LDS, 16B per lane, tracked by ASYNCcnt ----------
__device__ __forceinline__ void asyncLoadB128(unsigned ldsByteAddr, const void* gaddr) {
    asm volatile("global_load_async_to_lds_b128 %0, %1, off"
                 :: "v"(ldsByteAddr), "v"((unsigned long long)(uintptr_t)gaddr)
                 : "memory");
}
__device__ __forceinline__ void waitAsync0() {
    asm volatile("s_wait_asynccnt 0x0" ::: "memory");
}

// ---------- conversion: fp32 activations -> bf16 (optional ReLU, zero-pad K) ----------
__global__ void k_cvt_act(const float* __restrict__ src, int srcLd, int srcCols,
                          unsigned short* __restrict__ dst, int dstLd, int doRelu) {
    int m = blockIdx.y;
    int k = blockIdx.x * 256 + threadIdx.x;
    if (k >= dstLd) return;
    float v = (k < srcCols) ? src[(size_t)m * srcLd + k] : 0.0f;
    if (doRelu) v = fmaxf(v, 0.0f);
    dst[(size_t)m * dstLd + k] = f2bf(v);
}

// ---------- conversion: fp32 weight [K][N] -> bf16 transposed [Npad][Kpad], zero-padded ----------
__global__ void k_cvt_wt(const float* __restrict__ W, int K, int N,
                         unsigned short* __restrict__ Wt, int Kpad) {
    int n = blockIdx.y;
    int k = blockIdx.x * 256 + threadIdx.x;
    if (k >= Kpad) return;
    float v = (n < N && k < K) ? W[(size_t)k * N + n] : 0.0f;
    Wt[(size_t)n * Kpad + k] = f2bf(v);
}

// ---------- WMMA GEMM: C[M, ldOut] = A[M, Kpad](bf16) * Wt[Npad, Kpad]^T(bf16) + bias ----------
// Block: 256 threads (8 waves), tile 128(M) x 64(N), BK = 32, double-buffered LDS filled
// with global_load_async_to_lds_b128. Waves arranged 4(M) x 2(N); each wave computes a
// 32x32 tile via 2x2 v_wmma_f32_16x16x32_bf16.
#define LDA 40  /* halves per LDS row (32 + 8 pad); 80 bytes, 16B aligned */

// Stage one 128x32 A tile + 64x32 B tile into LDS buffer `buf` (async, branch-free).
// Out-of-range A rows are clamped to M-1: row r only feeds output row m0+r, which the
// epilogue masks, so clamped garbage never reaches a stored element.
__device__ __forceinline__ void stageTile(const unsigned short* __restrict__ A,
                                          const unsigned short* __restrict__ Wt,
                                          int M, int Kpad, int m0, int n0, int k0,
                                          int tid, unsigned asBase, unsigned bsBase) {
    #pragma unroll
    for (int c = 0; c < 2; ++c) {
        int chunk = tid + c * 256;
        int r  = chunk >> 2;            // 0..127
        int kc = (chunk & 3) << 3;      // 0,8,16,24 halves
        int gr = m0 + r; if (gr > M - 1) gr = M - 1;
        const unsigned short* gp = &A[(size_t)gr * Kpad + k0 + kc];
        asyncLoadB128(asBase + (unsigned)(r * (LDA * 2) + kc * 2), gp);
        __builtin_prefetch(gp + 64, 0, 1);          // global_prefetch (next-next K tile)
    }
    {
        int r  = tid >> 2;              // 0..63 (column of Wt)
        int kc = (tid & 3) << 3;
        const unsigned short* gp = &Wt[(size_t)(n0 + r) * Kpad + k0 + kc];
        asyncLoadB128(bsBase + (unsigned)(r * (LDA * 2) + kc * 2), gp);
        __builtin_prefetch(gp + 64, 0, 1);
    }
}

__global__ __launch_bounds__(256) void
k_gemm_bf16(const unsigned short* __restrict__ A,   // [M][Kpad] bf16 bits
            const unsigned short* __restrict__ Wt,  // [Npad][Kpad] bf16 bits (B transposed)
            const float* __restrict__ bias,         // [Nreal]
            float* __restrict__ C,                  // [M][ldOut]
            int M, int Kpad, int Nreal, int ldOut, int storeN, int doRelu) {
    __shared__ unsigned short As[2][128 * LDA];
    __shared__ unsigned short Bs[2][64 * LDA];

    const int tid      = threadIdx.x;
    const int lane     = tid & 31;
    const int wave     = tid >> 5;
    const int mw       = wave >> 1;   // 0..3
    const int nw       = wave & 1;    // 0..1
    const int laneRow  = lane & 15;
    const int laneHalf = lane >> 4;   // 0 or 1

    const int m0 = blockIdx.x * 128;
    const int n0 = blockIdx.y * 64;

    const unsigned asB[2] = { (unsigned)(uintptr_t)&As[0][0], (unsigned)(uintptr_t)&As[1][0] };
    const unsigned bsB[2] = { (unsigned)(uintptr_t)&Bs[0][0], (unsigned)(uintptr_t)&Bs[1][0] };

    floatx8 acc[2][2] = {};

    const int TILES = Kpad >> 5;

    // Prologue: fill buffer 0.
    stageTile(A, Wt, M, Kpad, m0, n0, 0, tid, asB[0], bsB[0]);
    waitAsync0();
    __syncthreads();

    int cur = 0;
    for (int t = 0; t < TILES; ++t) {
        const int nxt = cur ^ 1;
        // Issue next tile's async loads; they overlap with the WMMA work below.
        if (t + 1 < TILES)
            stageTile(A, Wt, M, Kpad, m0, n0, (t + 1) * 32, tid, asB[nxt], bsB[nxt]);

        // ---- fragments per ISA 16-bit A(16x32)/B(32x16) lane layouts ----
        const unsigned short* as = &As[cur][0];
        const unsigned short* bs = &Bs[cur][0];
        bf16x16 af[2], bfr[2];
        #pragma unroll
        for (int i = 0; i < 2; ++i) {
            int row = mw * 32 + i * 16 + laneRow;
            int klo = laneHalf * 8;   // lanes 0-15: K 0..7 / 16..23 ; lanes 16-31: K 8..15 / 24..31
            Bf16Pack p;
            p.lo = *(const ushort8*)&as[row * LDA + klo];
            p.hi = *(const ushort8*)&as[row * LDA + klo + 16];
            af[i] = __builtin_bit_cast(bf16x16, p);
        }
        #pragma unroll
        for (int j = 0; j < 2; ++j) {
            int col = nw * 32 + j * 16 + laneRow;
            int klo = laneHalf * 8;   // lanes 0-15: K 0..7 / 16..23 ; lanes 16-31: K 8..15 / 24..31
            Bf16Pack p;
            p.lo = *(const ushort8*)&bs[col * LDA + klo];
            p.hi = *(const ushort8*)&bs[col * LDA + klo + 16];
            bfr[j] = __builtin_bit_cast(bf16x16, p);
        }

        #pragma unroll
        for (int i = 0; i < 2; ++i)
            #pragma unroll
            for (int j = 0; j < 2; ++j)
                acc[i][j] = __builtin_amdgcn_wmma_f32_16x16x32_bf16(
                    false, af[i], false, bfr[j], (short)0, acc[i][j], false, false);

        if (t + 1 < TILES) {
            waitAsync0();      // next tile landed in LDS
            __syncthreads();   // all waves done reading `cur`, next tile visible to all
        }
        cur = nxt;
    }

    // ---- epilogue: bias (+ optional ReLU), per ISA C/D layout ----
    #pragma unroll
    for (int i = 0; i < 2; ++i) {
        #pragma unroll
        for (int j = 0; j < 2; ++j) {
            int gn = n0 + nw * 32 + j * 16 + laneRow;
            float bv = (gn < Nreal) ? bias[gn] : 0.0f;
            #pragma unroll
            for (int v = 0; v < 8; ++v) {
                int gr = m0 + mw * 32 + i * 16 + laneHalf * 8 + v;
                if (gr < M && gn < storeN) {
                    float o = acc[i][j][v] + bv;
                    if (doRelu) o = fmaxf(o, 0.0f);
                    C[(size_t)gr * ldOut + gn] = o;
                }
            }
        }
    }
}

// ---------- edge scatter-add: agg[dst[e]] += H[src[e]] (hardware f32 atomics) ----------
__global__ void k_scatter(const float* __restrict__ H, const int* __restrict__ src,
                          const int* __restrict__ dst, float* __restrict__ agg, int Npad) {
    int e = blockIdx.x;
    int c = blockIdx.y * 64 + threadIdx.x;
    int s = src[e], d = dst[e];
    float v = H[(size_t)s * Npad + c];
    __hip_atomic_fetch_add(&agg[(size_t)d * Npad + c], v,
                           __ATOMIC_RELAXED, __HIP_MEMORY_SCOPE_AGENT);
}

// ---------- host orchestration ----------
static inline size_t alignUp(size_t x) { return (x + 255) & ~(size_t)255; }

extern "C" void kernel_launch(void* const* d_in, const int* in_sizes, int n_in,
                              void* d_out, int out_size, void* d_ws, size_t ws_size,
                              hipStream_t stream) {
    const float* features = (const float*)d_in[0];
    const float* W1 = (const float*)d_in[1];  const float* b1 = (const float*)d_in[2];
    const float* W2 = (const float*)d_in[3];  const float* b2 = (const float*)d_in[4];
    const float* W3 = (const float*)d_in[5];  const float* b3 = (const float*)d_in[6];
    const float* Wi = (const float*)d_in[7];  const float* bi = (const float*)d_in[8];
    const int*   src = (const int*)d_in[9];
    const int*   dst = (const int*)d_in[10];

    const int M = N_NODES;
    const int nE = N_EDGES;

    // Layer dims (Kpad mult of 32, Npad mult of 64)
    const int K1 = 1433, Kp1 = 1440, N1 = 700, Np1 = 704;
    const int K2 = 700,  Kp2 = 704,  N2 = 400, Np2 = 448;
    const int K3 = 400,  Kp3 = 448,  N3 = 100, Np3 = 128;
    const int K4 = 100,  Kp4 = 128,  N4 = 16,  Np4 = 64;

    // Workspace carve-up
    char* base = (char*)d_ws;
    size_t off = 0;
    unsigned short* XB = (unsigned short*)(base + off); off += alignUp((size_t)M * Kp1 * 2); // 144.0 MB
    float*          H  = (float*)(base + off);          off += alignUp((size_t)M * Np1 * 4); // 140.8 MB
    float*          AG = (float*)(base + off);          off += alignUp((size_t)M * Np1 * 4); // 140.8 MB
    unsigned short* WT = (unsigned short*)(base + off); off += alignUp((size_t)Np1 * Kp1 * 2);
    (void)ws_size; (void)n_in; (void)in_sizes; (void)out_size;

    dim3 blk(256);

    // ===== Layer 1 =====
    k_cvt_act<<<dim3((Kp1 + 255) / 256, M), blk, 0, stream>>>(features, K1, K1, XB, Kp1, 0);
    k_cvt_wt <<<dim3((Kp1 + 255) / 256, Np1), blk, 0, stream>>>(W1, K1, N1, WT, Kp1);
    k_gemm_bf16<<<dim3((M + 127) / 128, Np1 / 64), blk, 0, stream>>>(XB, WT, b1, H, M, Kp1, N1, Np1, Np1, 0);
    hipMemsetAsync(AG, 0, (size_t)M * Np1 * 4, stream);
    k_scatter<<<dim3(nE, Np1 / 64), dim3(64), 0, stream>>>(H, src, dst, AG, Np1);

    // ===== Layer 2 =====
    k_cvt_act<<<dim3((Kp2 + 255) / 256, M), blk, 0, stream>>>(AG, Np1, Np1, XB, Kp2, 1); // ReLU
    k_cvt_wt <<<dim3((Kp2 + 255) / 256, Np2), blk, 0, stream>>>(W2, K2, N2, WT, Kp2);
    k_gemm_bf16<<<dim3((M + 127) / 128, Np2 / 64), blk, 0, stream>>>(XB, WT, b2, H, M, Kp2, N2, Np2, Np2, 0);
    hipMemsetAsync(AG, 0, (size_t)M * Np2 * 4, stream);
    k_scatter<<<dim3(nE, Np2 / 64), dim3(64), 0, stream>>>(H, src, dst, AG, Np2);

    // ===== Layer 3 =====
    k_cvt_act<<<dim3((Kp3 + 255) / 256, M), blk, 0, stream>>>(AG, Np2, Np2, XB, Kp3, 1); // ReLU
    k_cvt_wt <<<dim3((Kp3 + 255) / 256, Np3), blk, 0, stream>>>(W3, K3, N3, WT, Kp3);
    k_gemm_bf16<<<dim3((M + 127) / 128, Np3 / 64), blk, 0, stream>>>(XB, WT, b3, H, M, Kp3, N3, Np3, Np3, 0);
    hipMemsetAsync(AG, 0, (size_t)M * Np3 * 4, stream);
    k_scatter<<<dim3(nE, Np3 / 64), dim3(64), 0, stream>>>(H, src, dst, AG, Np3);

    // ===== Inclusion linear + final ReLU =====
    k_cvt_act<<<dim3((Kp4 + 255) / 256, M), blk, 0, stream>>>(AG, Np3, Np3, XB, Kp4, 0); // no ReLU here
    k_cvt_wt <<<dim3((Kp4 + 255) / 256, Np4), blk, 0, stream>>>(Wi, K4, N4, WT, Kp4);
    k_gemm_bf16<<<dim3((M + 127) / 128, Np4 / 64), blk, 0, stream>>>(
        XB, WT, bi, (float*)d_out, M, Kp4, N4, /*ldOut=*/16, /*storeN=*/16, /*relu=*/1);
}
